// JointModel_43430709297410
// MI455X (gfx1250) — matrix operand order repaired and checked by reading, compiled, not run on hardware
//
#include <hip/hip_runtime.h>

#define DEVFN __device__ __forceinline__

typedef _Float16 v16h __attribute__((ext_vector_type(16)));
typedef _Float16 h8   __attribute__((ext_vector_type(8)));
typedef float    v8f  __attribute__((ext_vector_type(8)));
typedef float    f4   __attribute__((ext_vector_type(4)));

constexpr int Bc   = 2;
constexpr int Tc   = 1096;
constexpr int HID  = 1024;
constexpr int FFN  = 4096;
constexpr int NH   = 8;
constexpr int HD   = 128;
constexpr int MALL = Bc * Tc;      // 2192 rows (joint stream)

constexpr int BM = 64, BN = 128, KC = 32;
constexpr int ASTR = 40;           // halves per A-tile LDS row (16B aligned, bank-padded)
constexpr int BSTR = 24;           // halves per B-frag lane slot (48B stride, 16B aligned)

DEVFN v8f zero8() { v8f z = {0.f,0.f,0.f,0.f,0.f,0.f,0.f,0.f}; return z; }
DEVFN h8  zeroh8(){ h8 z = {(_Float16)0,(_Float16)0,(_Float16)0,(_Float16)0,
                            (_Float16)0,(_Float16)0,(_Float16)0,(_Float16)0}; return z; }

DEVFN v8f wmma16(v16h a, v16h b, v8f c) {
  // v_wmma_f32_16x16x32_f16  (8 args: neg_a, A, neg_b, B, c_mod, C, reuse_a, reuse_b)
  return __builtin_amdgcn_wmma_f32_16x16x32_f16(false, a, false, b, (short)0, c, false, false);
}

// A-fragment (16x32 f16): lane holds row m = lane%16, K runs [g*16..g*16+15]
DEVFN v16h ld_fragA(const _Float16* rowp, int g) {
  h8 lo = *(const h8*)(rowp + g * 16);
  h8 hi = *(const h8*)(rowp + g * 16 + 8);
  v16h r;
#pragma unroll
  for (int i = 0; i < 8; ++i) { r[i] = lo[i]; r[i + 8] = hi[i]; }
  return r;
}

// B-fragment from LDS laid out [sub][g][n][BSTR]: lane reads 16 contiguous halves (K = g*16+i)
DEVFN v16h ld_fragB(const _Float16* Bs, int sub, int g, int n) {
  const _Float16* p = Bs + ((size_t)((sub * 2 + g) * 16 + n)) * BSTR;
  h8 lo = *(const h8*)p;
  h8 hi = *(const h8*)(p + 8);
  v16h r;
#pragma unroll
  for (int i = 0; i < 8; ++i) { r[i] = lo[i]; r[i + 8] = hi[i]; }
  return r;
}

// ---- cooperative tile loaders (256 threads) ----
DEVFN void load_atile(const _Float16* A, int lda, int M, int Le, int rowoff,
                      int m0, int kk0, int tid, _Float16* As) {
  int idx = tid * 8;                 // 2048 halves = 64x32
  int row = idx >> 5;
  int co  = idx & 31;
  int m   = m0 + row;
  int mm  = (m < M) ? m : (M - 1);
  int grow = (mm / Le) * Tc + rowoff + (mm % Le);
  h8 av = zeroh8();
  if (m < M) av = *(const h8*)(A + (size_t)grow * lda + kk0 + co);
  *(h8*)(As + row * ASTR + co) = av;
}

DEVFN void load_btile(const float* W, int N, int n0, int kk0, int tid, _Float16* Bs) {
  int idx = tid * 16;                // 4096 floats = 32x128, coalesced f32 reads
  int kr  = idx >> 7;                // K row 0..31
  int cb  = idx & 127;               // col offset, multiple of 16
  int sub = cb >> 4;
  const float* wp = W + (size_t)(kk0 + kr) * N + n0 + cb;
  int g = kr >> 4, i = kr & 15;
  _Float16* bs = Bs + ((size_t)((sub * 2 + g) * 16)) * BSTR + i;
#pragma unroll
  for (int j = 0; j < 16; j += 4) {
    f4 v = *(const f4*)(wp + j);
    bs[(j + 0) * BSTR] = (_Float16)v[0];
    bs[(j + 1) * BSTR] = (_Float16)v[1];
    bs[(j + 2) * BSTR] = (_Float16)v[2];
    bs[(j + 3) * BSTR] = (_Float16)v[3];
  }
}

// ================= GEMM: C[M,N] = A(f16)[M,K] * W(f32->f16)[K,N] =================
// EPI==0: store f16 into Cf16 (ldc).  EPI==1: Xres[grow*HID+n] += result (residual).
template<int EPI>
__global__ __launch_bounds__(256) void gemm_kernel(
    const _Float16* __restrict__ A, int lda,
    const float* __restrict__ W, int N, int K,
    int M, int Le, int rowoff,
    _Float16* __restrict__ Cf16, int ldc,
    float* __restrict__ Xres)
{
  __shared__ __align__(16) _Float16 As[BM * ASTR];
  __shared__ __align__(16) _Float16 Bs[8 * 2 * 16 * BSTR];
  int tid = threadIdx.x;
  int lane = tid & 31, wave = tid >> 5;
  int lm = lane & 15, lg = lane >> 4;
  int m0 = blockIdx.x * BM;
  int n0 = blockIdx.y * BN;
  int wm = (wave & 1) * 32;
  int wn = (wave >> 1) * 32;
  v8f acc[2][2];
#pragma unroll
  for (int a = 0; a < 2; ++a)
#pragma unroll
    for (int b = 0; b < 2; ++b) acc[a][b] = zero8();

  for (int kk0 = 0; kk0 < K; kk0 += KC) {
    load_atile(A, lda, M, Le, rowoff, m0, kk0, tid, As);
    load_btile(W, N, n0, kk0, tid, Bs);
    if (kk0 + 2 * KC <= K)   // prefetch next W tile -> global_prefetch_b8
      __builtin_prefetch(W + (size_t)(kk0 + KC + (tid >> 3)) * N + n0 + (tid & 7) * 16, 0, 3);
    __syncthreads();
    v16h a0 = ld_fragA(&As[(wm + lm) * ASTR], lg);
    v16h a1 = ld_fragA(&As[(wm + 16 + lm) * ASTR], lg);
    v16h b0 = ld_fragB(Bs, (wn >> 4) + 0, lg, lm);
    v16h b1 = ld_fragB(Bs, (wn >> 4) + 1, lg, lm);
    acc[0][0] = wmma16(a0, b0, acc[0][0]);
    acc[0][1] = wmma16(a0, b1, acc[0][1]);
    acc[1][0] = wmma16(a1, b0, acc[1][0]);
    acc[1][1] = wmma16(a1, b1, acc[1][1]);
    __syncthreads();
  }
#pragma unroll
  for (int ms = 0; ms < 2; ++ms)
#pragma unroll
    for (int ns = 0; ns < 2; ++ns)
#pragma unroll
      for (int r = 0; r < 8; ++r) {
        int m = m0 + wm + ms * 16 + r + 8 * lg;     // C layout: row r+8g, col lane%16
        if (m < M) {
          int grow = (m / Le) * Tc + rowoff + (m % Le);
          int n = n0 + wn + ns * 16 + lm;
          float v = acc[ms][ns][r];
          if (EPI == 0) Cf16[(size_t)grow * ldc + n] = (_Float16)v;
          else          Xres[(size_t)grow * HID + n] += v;
        }
      }
}

// ============== fused GeGLU: act = gelu_tanh(A*Wg) * (A*Wu), f16 out ==============
__global__ __launch_bounds__(256) void gemm_geglu_kernel(
    const _Float16* __restrict__ A, int lda,
    const float* __restrict__ Wg, const float* __restrict__ Wu,
    int N, int K, int M, int Le, int rowoff,
    _Float16* __restrict__ Out, int ldc)
{
  __shared__ __align__(16) _Float16 As[BM * ASTR];
  __shared__ __align__(16) _Float16 Bg[8 * 2 * 16 * BSTR];
  __shared__ __align__(16) _Float16 Bu[8 * 2 * 16 * BSTR];
  int tid = threadIdx.x;
  int lane = tid & 31, wave = tid >> 5;
  int lm = lane & 15, lg = lane >> 4;
  int m0 = blockIdx.x * BM, n0 = blockIdx.y * BN;
  int wm = (wave & 1) * 32, wn = (wave >> 1) * 32;
  v8f ag[2][2], au[2][2];
#pragma unroll
  for (int a = 0; a < 2; ++a)
#pragma unroll
    for (int b = 0; b < 2; ++b) { ag[a][b] = zero8(); au[a][b] = zero8(); }

  for (int kk0 = 0; kk0 < K; kk0 += KC) {
    load_atile(A, lda, M, Le, rowoff, m0, kk0, tid, As);
    load_btile(Wg, N, n0, kk0, tid, Bg);
    load_btile(Wu, N, n0, kk0, tid, Bu);
    __syncthreads();
    v16h a0  = ld_fragA(&As[(wm + lm) * ASTR], lg);
    v16h a1  = ld_fragA(&As[(wm + 16 + lm) * ASTR], lg);
    v16h bg0 = ld_fragB(Bg, (wn >> 4) + 0, lg, lm);
    v16h bg1 = ld_fragB(Bg, (wn >> 4) + 1, lg, lm);
    v16h bu0 = ld_fragB(Bu, (wn >> 4) + 0, lg, lm);
    v16h bu1 = ld_fragB(Bu, (wn >> 4) + 1, lg, lm);
    ag[0][0] = wmma16(a0, bg0, ag[0][0]); ag[0][1] = wmma16(a0, bg1, ag[0][1]);
    ag[1][0] = wmma16(a1, bg0, ag[1][0]); ag[1][1] = wmma16(a1, bg1, ag[1][1]);
    au[0][0] = wmma16(a0, bu0, au[0][0]); au[0][1] = wmma16(a0, bu1, au[0][1]);
    au[1][0] = wmma16(a1, bu0, au[1][0]); au[1][1] = wmma16(a1, bu1, au[1][1]);
    __syncthreads();
  }
#pragma unroll
  for (int ms = 0; ms < 2; ++ms)
#pragma unroll
    for (int ns = 0; ns < 2; ++ns)
#pragma unroll
      for (int r = 0; r < 8; ++r) {
        int m = m0 + wm + ms * 16 + r + 8 * lg;
        if (m < M) {
          int grow = (m / Le) * Tc + rowoff + (m % Le);
          int n = n0 + wn + ns * 16 + lm;
          float g = ag[ms][ns][r], u = au[ms][ns][r];
          float inner = 0.7978845608028654f * (g + 0.044715f * g * g * g);
          float gel = 0.5f * g * (1.f + tanhf(inner));
          Out[(size_t)grow * ldc + n] = (_Float16)(gel * u);
        }
      }
}

// ================= elementwise / norm kernels =================
__global__ __launch_bounds__(256) void scale_embed_kernel(
    const float* __restrict__ e0, const float* __restrict__ e1,
    const float* __restrict__ e2, float* __restrict__ X)
{
  int token = blockIdx.x;
  int b = token / Tc, t = token % Tc;
  const float* e; int L, off;
  if (t < 1024)      { e = e0; L = 1024; off = 0; }
  else if (t < 1032) { e = e1; L = 8;    off = 1024; }
  else               { e = e2; L = 64;   off = 1032; }
  const float* src = e + ((size_t)b * L + (t - off)) * HID;
  float* dst = X + (size_t)token * HID;
  int tid = threadIdx.x;
  f4 v = *(const f4*)(src + tid * 4);
  f4 o = { v[0] * 32.f, v[1] * 32.f, v[2] * 32.f, v[3] * 32.f };  // sqrt(1024)=32
  *(f4*)(dst + tid * 4) = o;
}

__global__ __launch_bounds__(256) void rmsnorm_kernel(
    const float* __restrict__ X,
    const float* __restrict__ w0, const float* __restrict__ w1, const float* __restrict__ w2,
    _Float16* __restrict__ Out)
{
  int token = blockIdx.x;
  int t = token % Tc;
  const float* w = (t < 1024) ? w0 : ((t < 1032) ? w1 : w2);
  const float* xp = X + (size_t)token * HID;
  int tid = threadIdx.x;
  f4 v = *(const f4*)(xp + tid * 4);
  float ss = v[0]*v[0] + v[1]*v[1] + v[2]*v[2] + v[3]*v[3];
#pragma unroll
  for (int d = 1; d < 32; d <<= 1) ss += __shfl_xor(ss, d, 32);
  __shared__ float red[8];
  if ((tid & 31) == 0) red[tid >> 5] = ss;
  __syncthreads();
  ss = red[0]+red[1]+red[2]+red[3]+red[4]+red[5]+red[6]+red[7];
  float rms = rsqrtf(ss * (1.0f / HID) + 1e-6f);
  f4 wv = *(const f4*)(w + tid * 4);
  _Float16* op = Out + (size_t)token * HID + tid * 4;
#pragma unroll
  for (int j = 0; j < 4; ++j) op[j] = (_Float16)(v[j] * rms * (1.f + wv[j]));
}

__global__ void rope_kernel(_Float16* __restrict__ X, int nheads,
                            const int* __restrict__ p0, const int* __restrict__ p1,
                            const int* __restrict__ p2)
{
  int idx = blockIdx.x * blockDim.x + threadIdx.x;
  int total = Bc * Tc * nheads * 64;
  if (idx >= total) return;
  int j = idx & 63;
  int r = idx >> 6;
  int hh = r % nheads; r /= nheads;
  int t = r % Tc; int b = r / Tc;
  const int* pp; int off, L;
  if (t < 1024)      { pp = p0; off = 0;    L = 1024; }
  else if (t < 1032) { pp = p1; off = 1024; L = 8; }
  else               { pp = p2; off = 1032; L = 64; }
  int pos = pp[b * L + (t - off)];
  float inv = powf(10000.f, -(float)(2 * j) * (1.f / 128.f));
  float ang = (float)pos * inv;
  float s, c; sincosf(ang, &s, &c);
  size_t base = ((size_t)(b * Tc + t)) * ((size_t)nheads * HD) + (size_t)hh * HD;
  float x1 = (float)X[base + j], x2 = (float)X[base + j + 64];
  X[base + j]      = (_Float16)(x1 * c - x2 * s);
  X[base + j + 64] = (_Float16)(x2 * c + x1 * s);
}

// ============== flash attention: 1 wave, 16 q-rows, stream kv in 32-chunks ==============
__global__ __launch_bounds__(32) void attn_kernel(
    const _Float16* __restrict__ Q, const _Float16* __restrict__ Kb,
    const _Float16* __restrict__ Vb, const float* __restrict__ mask,
    _Float16* __restrict__ O)
{
  __shared__ __align__(16) _Float16 Plds[16 * ASTR];
  __shared__ __align__(16) _Float16 Vlds[8 * 2 * 16 * BSTR];
  int lane = threadIdx.x;
  int lm = lane & 15, lg = lane >> 4;
  int qt = blockIdx.x, h = blockIdx.y, b = blockIdx.z;
  float m_i[8], l_i[8], scf[8], mx[8], ssum[8];
  v8f o[8];
#pragma unroll
  for (int r = 0; r < 8; ++r) { m_i[r] = -1e30f; l_i[r] = 0.f; }
#pragma unroll
  for (int sb = 0; sb < 8; ++sb) o[sb] = zero8();

  v16h qa[4];
  {
    int t = qt * 16 + lm; if (t >= Tc) t = Tc - 1;
    const _Float16* qp = Q + (size_t)(b * Tc + t) * (NH * HD) + (size_t)h * HD;
#pragma unroll
    for (int kk = 0; kk < 4; ++kk) qa[kk] = ld_fragA(qp + kk * 32, lg);
  }
  const float cl = 0.08838834764831845f / 50.f;  // (1/sqrt(128)) / softclamp
  const int NKCH = (Tc + 31) / 32;               // 35 kv chunks

  for (int kc = 0; kc < NKCH; ++kc) {
    int kb = kc * 32;
    asm volatile("s_wait_dscnt 0x0" ::: "memory");
    // stage V chunk (32 x 128) into B-frag order; lane owns one kv row
    {
      int kv = kb + lane;
      int kvc = (kv < Tc) ? kv : Tc - 1;
      const _Float16* vp = Vb + (size_t)(b * Tc + kvc) * HD;
      int g = lane >> 4, i = lane & 15;
#pragma unroll
      for (int d0 = 0; d0 < HD; d0 += 8) {
        h8 vv = *(const h8*)(vp + d0);
#pragma unroll
        for (int j = 0; j < 8; ++j) {
          int dd = d0 + j, sub = dd >> 4, n = dd & 15;
          Vlds[((size_t)((sub * 2 + g) * 16 + n)) * BSTR + i] =
              (kv < Tc) ? vv[j] : (_Float16)0.f;
        }
      }
    }
    // S = Q * K^T over HD (4 chunks of 32)
    v8f s0 = zero8(), s1 = zero8();
#pragma unroll
    for (int kk = 0; kk < 4; ++kk) {
      int kv0 = kb + lm;      int kv0c = (kv0 < Tc) ? kv0 : Tc - 1;
      int kv1 = kb + 16 + lm; int kv1c = (kv1 < Tc) ? kv1 : Tc - 1;
      const _Float16* kp0 = Kb + (size_t)(b * Tc + kv0c) * HD + kk * 32 + lg * 16;
      const _Float16* kp1 = Kb + (size_t)(b * Tc + kv1c) * HD + kk * 32 + lg * 16;
      h8 k0lo = *(const h8*)kp0; h8 k0hi = *(const h8*)(kp0 + 8);
      h8 k1lo = *(const h8*)kp1; h8 k1hi = *(const h8*)(kp1 + 8);
      v16h bk0, bk1;
#pragma unroll
      for (int i = 0; i < 8; ++i) {
        bk0[i] = k0lo[i]; bk0[i + 8] = k0hi[i];
        bk1[i] = k1lo[i]; bk1[i + 8] = k1hi[i];
      }
      s0 = wmma16(qa[kk], bk0, s0);
      s1 = wmma16(qa[kk], bk1, s1);
    }
    // softclamp + mask + online softmax
#pragma unroll
    for (int r = 0; r < 8; ++r) {
      int q = qt * 16 + r + 8 * lg; int qc = (q < Tc) ? q : Tc - 1;
      int k0 = kb + lm, k1 = kb + 16 + lm;
      float a0 = 50.f * tanhf(s0[r] * cl);
      float a1 = 50.f * tanhf(s1[r] * cl);
      const float* mrow = mask + ((size_t)b * Tc + qc) * Tc;
      a0 = (k0 < Tc) ? (a0 + mrow[k0]) : -1e30f;
      a1 = (k1 < Tc) ? (a1 + mrow[k1]) : -1e30f;
      s0[r] = a0; s1[r] = a1;
      mx[r] = fmaxf(a0, a1);
    }
#pragma unroll
    for (int d = 1; d < 16; d <<= 1)
#pragma unroll
      for (int r = 0; r < 8; ++r) mx[r] = fmaxf(mx[r], __shfl_xor(mx[r], d, 32));
#pragma unroll
    for (int r = 0; r < 8; ++r) {
      float mn = fmaxf(m_i[r], mx[r]);
      scf[r] = __expf(m_i[r] - mn);
      m_i[r] = mn;
      float p0 = __expf(s0[r] - mn);
      float p1 = __expf(s1[r] - mn);
      s0[r] = p0; s1[r] = p1;
      ssum[r] = p0 + p1;
    }
#pragma unroll
    for (int d = 1; d < 16; d <<= 1)
#pragma unroll
      for (int r = 0; r < 8; ++r) ssum[r] += __shfl_xor(ssum[r], d, 32);
#pragma unroll
    for (int r = 0; r < 8; ++r) l_i[r] = l_i[r] * scf[r] + ssum[r];
#pragma unroll
    for (int sb = 0; sb < 8; ++sb)
#pragma unroll
      for (int r = 0; r < 8; ++r) o[sb][r] *= scf[r];
    // P: C layout -> LDS -> A-frag layout
#pragma unroll
    for (int r = 0; r < 8; ++r) {
      int m = r + 8 * lg;
      Plds[m * ASTR + lm]      = (_Float16)s0[r];
      Plds[m * ASTR + 16 + lm] = (_Float16)s1[r];
    }
    asm volatile("s_wait_dscnt 0x0" ::: "memory");
    v16h pa = ld_fragA(&Plds[lm * ASTR], lg);
#pragma unroll
    for (int sb = 0; sb < 8; ++sb) {
      v16h bv = ld_fragB(Vlds, sb, lg, lm);
      o[sb] = wmma16(pa, bv, o[sb]);
    }
  }
#pragma unroll
  for (int sb = 0; sb < 8; ++sb)
#pragma unroll
    for (int r = 0; r < 8; ++r) {
      int t = qt * 16 + r + 8 * lg;
      if (t < Tc) {
        float val = o[sb][r] / l_i[r];
        O[(size_t)(b * Tc + t) * (NH * HD) + (size_t)h * HD + sb * 16 + lm] = (_Float16)val;
      }
    }
}

__global__ __launch_bounds__(256) void final_out_kernel(
    const float* __restrict__ X,
    const float* __restrict__ w0, const float* __restrict__ w1, const float* __restrict__ w2,
    float* __restrict__ Out)
{
  int token = blockIdx.x;
  int b = token / Tc, t = token % Tc;
  const float* w; size_t obase; int L, off;
  if (t < 1024)      { w = w0; L = 1024; off = 0;    obase = 0; }
  else if (t < 1032) { w = w1; L = 8;    off = 1024; obase = (size_t)Bc * 1024 * HID; }
  else               { w = w2; L = 64;   off = 1032; obase = (size_t)Bc * 1024 * HID + (size_t)Bc * 8 * HID; }
  const float* xp = X + (size_t)token * HID;
  int tid = threadIdx.x;
  f4 v = *(const f4*)(xp + tid * 4);
  float ss = v[0]*v[0] + v[1]*v[1] + v[2]*v[2] + v[3]*v[3];
#pragma unroll
  for (int d = 1; d < 32; d <<= 1) ss += __shfl_xor(ss, d, 32);
  __shared__ float red[8];
  if ((tid & 31) == 0) red[tid >> 5] = ss;
  __syncthreads();
  ss = red[0]+red[1]+red[2]+red[3]+red[4]+red[5]+red[6]+red[7];
  float rms = rsqrtf(ss * (1.0f / HID) + 1e-6f);
  f4 wv = *(const f4*)(w + tid * 4);
  float* op = Out + obase + ((size_t)b * L + (t - off)) * HID + tid * 4;
#pragma unroll
  for (int j = 0; j < 4; ++j) op[j] = v[j] * rms * (1.f + wv[j]);
}

// ======================= host launcher =======================
extern "C" void kernel_launch(void* const* d_in, const int* in_sizes, int n_in,
                              void* d_out, int out_size, void* d_ws, size_t ws_size,
                              hipStream_t stream)
{
  const float *emb[3]; const int *pos[3]; const float *mask;
  const float *wq[3][4], *wk[3][4], *wvp[3][4], *wo[3][4];
  const float *inln[3][4], *postln[3][4], *gate[3][4], *up[3][4], *down[3][4];
  const float *finln[3];

  bool sorted = (n_in > 0 && in_sizes[0] == Bc * Tc * Tc);   // mask first => sorted pytree order
  if (!sorted) {
    // setup_inputs() insertion order
    emb[0] = (const float*)d_in[0]; pos[0] = (const int*)d_in[1];
    emb[1] = (const float*)d_in[2]; pos[1] = (const int*)d_in[3];
    emb[2] = (const float*)d_in[4]; pos[2] = (const int*)d_in[5];
    mask   = (const float*)d_in[6];
    for (int e = 0; e < 3; ++e) {
      int base = 7 + e * 37;
      for (int l = 0; l < 4; ++l) {
        int lb = base + l * 9;
        wq[e][l]     = (const float*)d_in[lb + 0];
        wk[e][l]     = (const float*)d_in[lb + 1];
        wvp[e][l]    = (const float*)d_in[lb + 2];
        wo[e][l]     = (const float*)d_in[lb + 3];
        inln[e][l]   = (const float*)d_in[lb + 4];
        postln[e][l] = (const float*)d_in[lb + 5];
        gate[e][l]   = (const float*)d_in[lb + 6];
        up[e][l]     = (const float*)d_in[lb + 7];
        down[e][l]   = (const float*)d_in[lb + 8];
      }
      finln[e] = (const float*)d_in[base + 36];
    }
  } else {
    // sorted: attention_mask, emb_action, emb_proprio, emb_vlm, params(sorted), pos_action, pos_proprio, pos_vlm
    mask   = (const float*)d_in[0];
    emb[2] = (const float*)d_in[1];
    emb[1] = (const float*)d_in[2];
    emb[0] = (const float*)d_in[3];
    int bases[3] = {78, 41, 4};      // vlm, proprio, action (experts sorted a<p<v)
    for (int e = 0; e < 3; ++e) {
      int base = bases[e];
      finln[e] = (const float*)d_in[base];
      for (int l = 0; l < 4; ++l) {
        int lb = base + 1 + l * 9;   // layer dict sorted: down,gate,in_ln,post_ln,up,wk,wo,wq,wv
        down[e][l]   = (const float*)d_in[lb + 0];
        gate[e][l]   = (const float*)d_in[lb + 1];
        inln[e][l]   = (const float*)d_in[lb + 2];
        postln[e][l] = (const float*)d_in[lb + 3];
        up[e][l]     = (const float*)d_in[lb + 4];
        wk[e][l]     = (const float*)d_in[lb + 5];
        wo[e][l]     = (const float*)d_in[lb + 6];
        wq[e][l]     = (const float*)d_in[lb + 7];
        wvp[e][l]    = (const float*)d_in[lb + 8];
      }
    }
    pos[2] = (const int*)d_in[115];
    pos[1] = (const int*)d_in[116];
    pos[0] = (const int*)d_in[117];
  }

  auto align_up = [](size_t v) { return (v + 255) & ~(size_t)255; };
  char* p = (char*)d_ws;
  float*    x   = (float*)p;    p += align_up((size_t)MALL * HID * 4);
  _Float16* hN  = (_Float16*)p; p += align_up((size_t)MALL * HID * 2);
  _Float16* Qr  = (_Float16*)p; p += align_up((size_t)MALL * HID * 2);
  _Float16* Kr  = (_Float16*)p; p += align_up((size_t)MALL * HD * 2);
  _Float16* Vv  = (_Float16*)p; p += align_up((size_t)MALL * HD * 2);
  _Float16* aO  = (_Float16*)p; p += align_up((size_t)MALL * HID * 2);
  _Float16* act = (_Float16*)p; p += align_up((size_t)MALL * FFN * 2);
  (void)ws_size; (void)out_size;

  int Ms[3]   = {Bc * 1024, Bc * 8, Bc * 64};
  int Ls[3]   = {1024, 8, 64};
  int offs[3] = {0, 1024, 1032};

  scale_embed_kernel<<<MALL, 256, 0, stream>>>(emb[0], emb[1], emb[2], x);

  for (int l = 0; l < 4; ++l) {
    rmsnorm_kernel<<<MALL, 256, 0, stream>>>(x, inln[0][l], inln[1][l], inln[2][l], hN);
    for (int e = 0; e < 3; ++e) {
      dim3 gq((Ms[e] + BM - 1) / BM, HID / BN);
      gemm_kernel<0><<<gq, 256, 0, stream>>>(hN, HID, wq[e][l], HID, HID,
                                             Ms[e], Ls[e], offs[e], Qr, HID, nullptr);
      dim3 gk((Ms[e] + BM - 1) / BM, HD / BN);
      gemm_kernel<0><<<gk, 256, 0, stream>>>(hN, HID, wk[e][l], HD, HID,
                                             Ms[e], Ls[e], offs[e], Kr, HD, nullptr);
      gemm_kernel<0><<<gk, 256, 0, stream>>>(hN, HID, wvp[e][l], HD, HID,
                                             Ms[e], Ls[e], offs[e], Vv, HD, nullptr);
    }
    {
      int tot = Bc * Tc * NH * 64;
      rope_kernel<<<(tot + 255) / 256, 256, 0, stream>>>(Qr, NH, pos[0], pos[1], pos[2]);
      tot = Bc * Tc * 64;
      rope_kernel<<<(tot + 255) / 256, 256, 0, stream>>>(Kr, 1, pos[0], pos[1], pos[2]);
    }
    dim3 ga((Tc + 15) / 16, NH, Bc);
    attn_kernel<<<ga, 32, 0, stream>>>(Qr, Kr, Vv, mask, aO);
    for (int e = 0; e < 3; ++e) {
      dim3 go((Ms[e] + BM - 1) / BM, HID / BN);
      gemm_kernel<1><<<go, 256, 0, stream>>>(aO, HID, wo[e][l], HID, HID,
                                             Ms[e], Ls[e], offs[e], nullptr, 0, x);
    }
    rmsnorm_kernel<<<MALL, 256, 0, stream>>>(x, postln[0][l], postln[1][l], postln[2][l], hN);
    for (int e = 0; e < 3; ++e) {
      dim3 gg((Ms[e] + BM - 1) / BM, FFN / BN);
      gemm_geglu_kernel<<<gg, 256, 0, stream>>>(hN, HID, gate[e][l], up[e][l], FFN, HID,
                                                Ms[e], Ls[e], offs[e], act, FFN);
      dim3 gd((Ms[e] + BM - 1) / BM, HID / BN);
      gemm_kernel<1><<<gd, 256, 0, stream>>>(act, FFN, down[e][l], HID, FFN,
                                             Ms[e], Ls[e], offs[e], nullptr, 0, x);
    }
  }
  final_out_kernel<<<MALL, 256, 0, stream>>>(x, finln[0], finln[1], finln[2], (float*)d_out);
}
